// NunchakuQwenImageTransformer2DModel_28905129902662
// MI455X (gfx1250) — matrix operand order repaired and checked by reading, not compiled, not run on hardware
//
#include <hip/hip_runtime.h>
#include <cstddef>

// ---------------------------------------------------------------------------
// Types for CDNA5 WMMA (gfx1250, wave32): v16bf A/B fragments, v8f accumulator
// ---------------------------------------------------------------------------
typedef __attribute__((ext_vector_type(16))) __bf16 v16bf;
typedef __attribute__((ext_vector_type(8)))  float  v8f;
typedef int v4i __attribute__((vector_size(16)));   // matches async-LDS builtin param

struct alignas(16) U128 { unsigned a0, a1, a2, a3; };
struct alignas(8)  U64  { unsigned lo, hi; };

union Frag { U128 q[2]; v16bf v; };

// CDNA5 async global->LDS copy (ASYNCcnt), guarded so either toolchain compiles
#if defined(__has_builtin)
#if __has_builtin(__builtin_amdgcn_global_load_async_to_lds_b128) && \
    __has_builtin(__builtin_amdgcn_s_wait_asynccnt)
#define HAVE_ASYNC_LDS 1
#endif
#endif
#ifndef HAVE_ASYNC_LDS
#define HAVE_ASYNC_LDS 0
#endif

__device__ __forceinline__ unsigned short f2bf(float f) {
  unsigned u = __float_as_uint(f);
  u += 0x7fffu + ((u >> 16) & 1u);          // round-to-nearest-even
  return (unsigned short)(u >> 16);
}

// pack two floats into one dword of bf16 pairs (RNE)
__device__ __forceinline__ unsigned pack_bf2(float a, float b) {
  unsigned ua = __float_as_uint(a);
  ua += 0x7fffu + ((ua >> 16) & 1u);
  unsigned ub = __float_as_uint(b);
  ub += 0x7fffu + ((ub >> 16) & 1u);
  return (ua >> 16) | (ub & 0xffff0000u);
}

__device__ __forceinline__ v8f vzero8() {
  v8f z = {0.f, 0.f, 0.f, 0.f, 0.f, 0.f, 0.f, 0.f};
  return z;
}

__device__ __forceinline__ float gelu_tanh(float x) {
  float x3 = x * x * x;
  return 0.5f * x * (1.f + tanhf(0.7978845608028654f * (x + 0.044715f * x3)));
}

#define DIM   3072
#define HEADS 24
#define HD    128
#define SEQ   1536
#define STXT  512
#define SIMG  1024

// ---------------------------------------------------------------------------
// silu(temb) -> s[3072]
// ---------------------------------------------------------------------------
__global__ void silu_kernel(const float* __restrict__ t, float* __restrict__ s) {
  int i = blockIdx.x * 256 + threadIdx.x;
  if (i < DIM) { float v = t[i]; s[i] = v / (1.f + __expf(-v)); }
}

// ---------------------------------------------------------------------------
// modulation GEMV: out_reordered[j*3072+i] = dot(s, W[:, i*6+j]) + b[i*6+j]
// ---------------------------------------------------------------------------
__global__ void __launch_bounds__(256)
modv_kernel(const float* __restrict__ s, const float* __restrict__ W,
            const float* __restrict__ b, float* __restrict__ out) {
  int c = blockIdx.x * 256 + threadIdx.x;     // 0..18431
  float acc = 0.f;
  #pragma unroll 4
  for (int k = 0; k < DIM; ++k)
    acc = fmaf(s[k], W[(size_t)k * (6 * DIM) + c], acc);
  acc += b[c];
  int i = c / 6, j = c % 6;
  out[(size_t)j * DIM + i] = acc;
}

// ---------------------------------------------------------------------------
// LayerNorm (no affine) + modulate -> bf16 activations (A-ready, row-major)
// ---------------------------------------------------------------------------
__global__ void __launch_bounds__(256)
ln_mod_kernel(const float* __restrict__ x, const float* __restrict__ shift,
              const float* __restrict__ scale, unsigned short* __restrict__ out) {
  const int row = blockIdx.x;
  const int t = threadIdx.x;
  const float* xr = x + (size_t)row * DIM;
  __shared__ float redA[256];
  __shared__ float redB[256];
  float s = 0.f, s2 = 0.f;
  for (int i = t; i < DIM; i += 256) { float v = xr[i]; s += v; s2 += v * v; }
  redA[t] = s; redB[t] = s2; __syncthreads();
  for (int st = 128; st > 0; st >>= 1) {
    if (t < st) { redA[t] += redA[t + st]; redB[t] += redB[t + st]; }
    __syncthreads();
  }
  float mu  = redA[0] * (1.f / DIM);
  float var = redB[0] * (1.f / DIM) - mu * mu;
  float inv = rsqrtf(var + 1e-6f);
  unsigned short* orow = out + (size_t)row * DIM;
  for (int i = t; i < DIM; i += 256)
    orow[i] = f2bf((xr[i] - mu) * inv * (scale[i] + 1.f) + shift[i]);
}

// ---------------------------------------------------------------------------
// Tiled WMMA GEMM: C[M,N] = epi(A_bf16[M,K] @ W_f32[K,N] + bias)
// Block 128x256x32, 256 threads = 8 waves (2 x 4), wave tile 64x64:
// 16 WMMA per K-step per wave fed by 16 ds_load_b128 (1.0 LDS op / WMMA).
// Double-buffered LDS, software pipelined: tile i+1 staged (async A copy via
// GLOBAL_LOAD_ASYNC_TO_LDS; W fp32->bf16 packed in registers) while WMMAs
// for tile i run. One barrier per K-step.
// epi: 0 = f32 store, 1 = bf16 gelu store, 2 = f32 res + gate*val store
// ---------------------------------------------------------------------------
#define BM  128
#define BN  256
#define BK  32
#define BKP 40

__global__ void __launch_bounds__(256)
gemm_bf16_kernel(const unsigned short* __restrict__ A, const float* __restrict__ W,
                 const float* __restrict__ bias, int M, int N, int K, int epi,
                 float* __restrict__ outF, unsigned short* __restrict__ outB,
                 const float* __restrict__ res, const float* __restrict__ gate) {
  __shared__ unsigned short As[2][BM * BKP];
  __shared__ unsigned short Bs[2][BN * BKP];
  const int t    = threadIdx.x;
  const int lane = t & 31;
  const int wid  = t >> 5;
  const int wm   = (wid & 1) * 64;     // 2 waves along M
  const int wn   = (wid >> 1) * 64;    // 4 waves along N
  const int bm0  = blockIdx.y * BM;
  const int bn0  = blockIdx.x * BN;
  const int l16  = lane & 15;
  const int hi   = lane >> 4;
  // A-copy geometry: 512 b128 chunks; chunk c -> row c>>2, kcol (c&3)*8
  const int ar0 = t >> 2;
  const int ar1 = (t + 256) >> 2;
  const int akc = (t & 3) << 3;

  // B-copy geometry: thread owns one N column (t), 8 groups of 4 K-contiguous
  float breg[32];
  auto loadB = [&](int k0) {
    #pragma unroll
    for (int i = 0; i < 8; ++i) {
      int kk = k0 + i * 4;
      #pragma unroll
      for (int j = 0; j < 4; ++j)
        breg[i * 4 + j] = W[(size_t)(kk + j) * N + bn0 + t];
    }
  };
  auto storeB = [&](int buf) {
    #pragma unroll
    for (int i = 0; i < 8; ++i) {
      U64 u;
      u.lo = pack_bf2(breg[i * 4 + 0], breg[i * 4 + 1]);
      u.hi = pack_bf2(breg[i * 4 + 2], breg[i * 4 + 3]);
      *(U64*)(&Bs[buf][t * BKP + i * 4]) = u;
    }
  };

#if HAVE_ASYNC_LDS
  auto loadA = [&](int k0, int buf) {
    __builtin_amdgcn_global_load_async_to_lds_b128(
        (v4i*)const_cast<unsigned short*>(A + (size_t)(bm0 + ar0) * K + k0 + akc),
        (v4i*)&As[buf][ar0 * BKP + akc], 0, 0);
    __builtin_amdgcn_global_load_async_to_lds_b128(
        (v4i*)const_cast<unsigned short*>(A + (size_t)(bm0 + ar1) * K + k0 + akc),
        (v4i*)&As[buf][ar1 * BKP + akc], 0, 0);
  };
#else
  U128 areg[2];
  auto loadA = [&](int k0, int) {
    areg[0] = *(const U128*)(A + (size_t)(bm0 + ar0) * K + k0 + akc);
    areg[1] = *(const U128*)(A + (size_t)(bm0 + ar1) * K + k0 + akc);
  };
  auto storeA = [&](int buf) {
    *(U128*)(&As[buf][ar0 * BKP + akc]) = areg[0];
    *(U128*)(&As[buf][ar1 * BKP + akc]) = areg[1];
  };
#endif

  v8f acc[4][4];
  #pragma unroll
  for (int mt = 0; mt < 4; ++mt)
    #pragma unroll
    for (int nt = 0; nt < 4; ++nt) acc[mt][nt] = vzero8();

  // prologue: stage tile 0
  loadB(0);
  loadA(0, 0);

  int buf = 0;
  for (int k0 = 0; k0 < K; k0 += BK) {
    storeB(buf);
#if !HAVE_ASYNC_LDS
    storeA(buf);
#endif
#if HAVE_ASYNC_LDS
    __builtin_amdgcn_s_wait_asynccnt(0);   // this wave's A tile landed in LDS
#endif
    __syncthreads();                        // publish tile; fence prior reads
    if (k0 + BK < K) {                      // stage tile i+1 into other buffer
      loadA(k0 + BK, buf ^ 1);
      loadB(k0 + BK);
    }

    // ---- fragments ----
    v16bf af[4];
    #pragma unroll
    for (int mt = 0; mt < 4; ++mt) {
      Frag a;
      int r  = wm + mt * 16 + l16;
      int kb = hi * 8;
      a.q[0] = *(const U128*)(&As[buf][r * BKP + kb]);
      a.q[1] = *(const U128*)(&As[buf][r * BKP + kb + 16]);
      af[mt] = a.v;
    }
    v16bf bfr[4];
    #pragma unroll
    for (int nt = 0; nt < 4; ++nt) {
      Frag b;
      int cc = wn + nt * 16 + l16;
      int kb = hi * 8;
      b.q[0] = *(const U128*)(&Bs[buf][cc * BKP + kb]);
      b.q[1] = *(const U128*)(&Bs[buf][cc * BKP + kb + 16]);
      bfr[nt] = b.v;
    }
    #pragma unroll
    for (int mt = 0; mt < 4; ++mt)
      #pragma unroll
      for (int nt = 0; nt < 4; ++nt)
        acc[mt][nt] = __builtin_amdgcn_wmma_f32_16x16x32_bf16(
            false, af[mt], false, bfr[nt], (short)0, acc[mt][nt], false, false);
    buf ^= 1;
  }

  // ---- epilogue (C layout: VGPR r -> rows {r, r+8}; lane%16 -> column) ----
  #pragma unroll
  for (int mt = 0; mt < 4; ++mt)
    #pragma unroll
    for (int nt = 0; nt < 4; ++nt)
      #pragma unroll
      for (int r = 0; r < 8; ++r) {
        int m = bm0 + wm + mt * 16 + r + hi * 8;
        int n = bn0 + wn + nt * 16 + l16;
        float v = acc[mt][nt][r] + bias[n];
        size_t idx = (size_t)m * N + n;
        if (epi == 0)      outF[idx] = v;
        else if (epi == 1) outB[idx] = f2bf(gelu_tanh(v));
        else               outF[idx] = res[idx] + gate[n] * v;
      }
}

// ---------------------------------------------------------------------------
// RMSNorm (per head-dim) + RoPE + 1/sqrt(HD) on Q -> head-major bf16 Q/K/V
// ---------------------------------------------------------------------------
__global__ void __launch_bounds__(64)
rms_rope_kernel(const float* __restrict__ qkv,
                const float* __restrict__ nqw, const float* __restrict__ nkw,
                const float* __restrict__ naqw, const float* __restrict__ nakw,
                const float* __restrict__ cosb, const float* __restrict__ sinb,
                unsigned short* __restrict__ Q, unsigned short* __restrict__ Kh,
                unsigned short* __restrict__ V) {
  const int sh = blockIdx.x;
  const int s = sh / HEADS, h = sh % HEADS;
  const int t = threadIdx.x;
  const int d0 = 2 * t, d1 = 2 * t + 1;
  const float* row = qkv + (size_t)s * (3 * DIM) + h * HD;
  float q0 = row[d0],            q1 = row[d1];
  float k0 = row[DIM + d0],      k1 = row[DIM + d1];
  float v0 = row[2 * DIM + d0],  v1 = row[2 * DIM + d1];

  __shared__ float redQ[64];
  __shared__ float redK[64];
  redQ[t] = q0 * q0 + q1 * q1;
  redK[t] = k0 * k0 + k1 * k1;
  __syncthreads();
  for (int st = 32; st > 0; st >>= 1) {
    if (t < st) { redQ[t] += redQ[t + st]; redK[t] += redK[t + st]; }
    __syncthreads();
  }
  const float* qw = (s < STXT) ? naqw : nqw;
  const float* kw = (s < STXT) ? nakw : nkw;
  float qi = rsqrtf(redQ[0] * (1.f / HD) + 1e-6f);
  float ki = rsqrtf(redK[0] * (1.f / HD) + 1e-6f);
  q0 *= qi * qw[d0]; q1 *= qi * qw[d1];
  k0 *= ki * kw[d0]; k1 *= ki * kw[d1];

  float c0 = cosb[(size_t)s * HD + d0], c1 = cosb[(size_t)s * HD + d1];
  float s0 = sinb[(size_t)s * HD + d0], s1 = sinb[(size_t)s * HD + d1];
  float qr0 = q0 * c0 - q1 * s0, qr1 = q1 * c1 + q0 * s1;
  float kr0 = k0 * c0 - k1 * s0, kr1 = k1 * c1 + k0 * s1;
  const float sc = 0.08838834764831845f;  // 1/sqrt(128), folded into Q
  size_t o = ((size_t)h * SEQ + s) * HD;
  Q[o + d0] = f2bf(qr0 * sc); Q[o + d1] = f2bf(qr1 * sc);
  Kh[o + d0] = f2bf(kr0);     Kh[o + d1] = f2bf(kr1);
  V[o + d0] = f2bf(v0);       V[o + d1] = f2bf(v1);
}

// ---------------------------------------------------------------------------
// Flash attention: block = (head, 64 q rows), 4 waves, wave owns 16 q rows.
// QK^T via WMMA, online softmax (16-lane butterflies), P staged through LDS
// (C->A layout flip), P@V via WMMA with V transposed in LDS.
// ---------------------------------------------------------------------------
__global__ void __launch_bounds__(128)
attn_kernel(const unsigned short* __restrict__ Q, const unsigned short* __restrict__ K,
            const unsigned short* __restrict__ V, unsigned short* __restrict__ O) {
  const int h  = blockIdx.y;
  const int qb = blockIdx.x * 64;
  const int t = threadIdx.x, lane = t & 31, w = t >> 5;
  const int l16 = lane & 15, hi = lane >> 4;
  const unsigned short* Qh = Q + (size_t)h * SEQ * HD;
  const unsigned short* Kp = K + (size_t)h * SEQ * HD;
  const unsigned short* Vp = V + (size_t)h * SEQ * HD;

  __shared__ unsigned short Qs[64 * 136];
  __shared__ unsigned short Ks[32 * 136];
  __shared__ unsigned short Vt[128 * 40];
  __shared__ unsigned short Ps[4 * 16 * 40];

  #pragma unroll
  for (int i = 0; i < 8; ++i) {   // 64x128 bf16 Q tile
    int c = t + i * 128;
    int r = c >> 4, kc = (c & 15) << 3;
    *(U128*)(Qs + r * 136 + kc) = *(const U128*)(Qh + (size_t)(qb + r) * HD + kc);
  }

  float mrun[8], lrun[8];
  v8f o[8];
  #pragma unroll
  for (int r = 0; r < 8; ++r) { mrun[r] = -1e30f; lrun[r] = 0.f; }
  #pragma unroll
  for (int nt = 0; nt < 8; ++nt) o[nt] = vzero8();

  for (int kc = 0; kc < SEQ; kc += 32) {
    __syncthreads();
    {  // K chunk: batch loads into temps, then store (no per-chunk stalls)
      U128 kreg[4];
      #pragma unroll
      for (int i = 0; i < 4; ++i) {
        int c = t + i * 128;
        int r = c >> 4, kcol = (c & 15) << 3;
        kreg[i] = *(const U128*)(Kp + (size_t)(kc + r) * HD + kcol);
      }
      #pragma unroll
      for (int i = 0; i < 4; ++i) {
        int c = t + i * 128;
        int r = c >> 4, kcol = (c & 15) << 3;
        *(U128*)(Ks + r * 136 + kcol) = kreg[i];
      }
    }
    {  // V chunk transposed [d][kv]: batch the strided loads, then pack+store
      int d = t;
      unsigned vv[32];
      #pragma unroll
      for (int p = 0; p < 32; ++p)
        vv[p] = Vp[(size_t)(kc + p) * HD + d];
      #pragma unroll
      for (int p = 0; p < 16; ++p)
        *(unsigned*)(Vt + d * 40 + 2 * p) = vv[2 * p] | (vv[2 * p + 1] << 16);
    }
    __syncthreads();

    // ---- scores: 16 q rows x 32 kv cols, gemm-K = 128 (4 WMMA steps) ----
    v8f st[2];
    st[0] = vzero8(); st[1] = vzero8();
    #pragma unroll
    for (int ks = 0; ks < 4; ++ks) {
      Frag a;
      int r  = w * 16 + l16;
      int kb = ks * 32 + hi * 8;
      a.q[0] = *(const U128*)(Qs + r * 136 + kb);
      a.q[1] = *(const U128*)(Qs + r * 136 + kb + 16);
      #pragma unroll
      for (int nt = 0; nt < 2; ++nt) {
        Frag b;
        int col = nt * 16 + l16;
        int kb2 = ks * 32 + hi * 8;
        b.q[0] = *(const U128*)(Ks + col * 136 + kb2);
        b.q[1] = *(const U128*)(Ks + col * 136 + kb2 + 16);
        st[nt] = __builtin_amdgcn_wmma_f32_16x16x32_bf16(
            false, a.v, false, b.v, (short)0, st[nt], false, false);
      }
    }

    // ---- online softmax (rows split across lane halves; xor<=8 stays in half)
    #pragma unroll
    for (int r = 0; r < 8; ++r) {
      float v = fmaxf(st[0][r], st[1][r]);
      #pragma unroll
      for (int off = 8; off > 0; off >>= 1) v = fmaxf(v, __shfl_xor(v, off, 32));
      float mnew = fmaxf(mrun[r], v);
      float corr = __expf(mrun[r] - mnew);
      float p0 = __expf(st[0][r] - mnew);
      float p1 = __expf(st[1][r] - mnew);
      float rs = p0 + p1;
      #pragma unroll
      for (int off = 8; off > 0; off >>= 1) rs += __shfl_xor(rs, off, 32);
      lrun[r] = lrun[r] * corr + rs;
      mrun[r] = mnew;
      #pragma unroll
      for (int nt = 0; nt < 8; ++nt) o[nt][r] *= corr;
      int m = r + hi * 8;
      Ps[(w * 16 + m) * 40 + l16]      = f2bf(p0);
      Ps[(w * 16 + m) * 40 + 16 + l16] = f2bf(p1);
    }

    // ---- P @ V: gemm-K = 32, one WMMA per 16-wide d tile (8 tiles) ----
    Frag a;
    int kb = hi * 8;
    a.q[0] = *(const U128*)(Ps + (w * 16 + l16) * 40 + kb);
    a.q[1] = *(const U128*)(Ps + (w * 16 + l16) * 40 + kb + 16);
    #pragma unroll
    for (int nt = 0; nt < 8; ++nt) {
      Frag b;
      int d   = nt * 16 + l16;
      int kb2 = hi * 8;
      b.q[0] = *(const U128*)(Vt + d * 40 + kb2);
      b.q[1] = *(const U128*)(Vt + d * 40 + kb2 + 16);
      o[nt] = __builtin_amdgcn_wmma_f32_16x16x32_bf16(
          false, a.v, false, b.v, (short)0, o[nt], false, false);
    }
  }

  #pragma unroll
  for (int nt = 0; nt < 8; ++nt)
    #pragma unroll
    for (int r = 0; r < 8; ++r) {
      int m = qb + w * 16 + r + hi * 8;
      int d = nt * 16 + l16;
      O[(size_t)m * DIM + h * HD + d] = f2bf(o[nt][r] / lrun[r]);
    }
}

// ---------------------------------------------------------------------------
// Host-side orchestration (all launches on `stream`; scratch in d_ws)
// ---------------------------------------------------------------------------
extern "C" void kernel_launch(void* const* d_in, const int* in_sizes, int n_in,
                              void* d_out, int out_size, void* d_ws, size_t ws_size,
                              hipStream_t stream) {
  (void)in_sizes; (void)n_in; (void)out_size; (void)ws_size;
  const float* hs    = (const float*)d_in[0];   // hidden_states [1024,3072]
  const float* es    = (const float*)d_in[1];   // encoder_hidden_states [512,3072]
  const float* temb  = (const float*)d_in[3];
  const float* rcos  = (const float*)d_in[4];
  const float* rsin  = (const float*)d_in[5];
  const float* w_imod = (const float*)d_in[6];  const float* b_imod = (const float*)d_in[7];
  const float* w_tmod = (const float*)d_in[8];  const float* b_tmod = (const float*)d_in[9];
  const float* nqw  = (const float*)d_in[10];   const float* nkw  = (const float*)d_in[11];
  const float* naqw = (const float*)d_in[12];   const float* nakw = (const float*)d_in[13];
  const float* w_qkv  = (const float*)d_in[14]; const float* b_qkv  = (const float*)d_in[15];
  const float* w_aqkv = (const float*)d_in[16]; const float* b_aqkv = (const float*)d_in[17];
  const float* w_o    = (const float*)d_in[18]; const float* b_o    = (const float*)d_in[19];
  const float* w_ao   = (const float*)d_in[20]; const float* b_ao   = (const float*)d_in[21];
  const float* w_im1  = (const float*)d_in[22]; const float* b_im1  = (const float*)d_in[23];
  const float* w_im2  = (const float*)d_in[24]; const float* b_im2  = (const float*)d_in[25];
  const float* w_tm1  = (const float*)d_in[26]; const float* b_tm1  = (const float*)d_in[27];
  const float* w_tm2  = (const float*)d_in[28]; const float* b_tm2  = (const float*)d_in[29];
  float* out = (float*)d_out;   // enc [512*3072] then hidden [1024*3072]

  char* ws = (char*)d_ws;
  size_t off = 0;
  auto alloc = [&](size_t b) { size_t o = off; off += (b + 255) & ~(size_t)255; return o; };

  float* s_temb  = (float*)(ws + alloc((size_t)DIM * 4));
  float* img_mod = (float*)(ws + alloc((size_t)6 * DIM * 4));
  float* txt_mod = (float*)(ws + alloc((size_t)6 * DIM * 4));
  unsigned short* A_img = (unsigned short*)(ws + alloc((size_t)SIMG * DIM * 2));
  unsigned short* A_txt = (unsigned short*)(ws + alloc((size_t)STXT * DIM * 2));
  size_t qkv_off = alloc((size_t)SEQ * 3 * DIM * 4);
  float* qkv = (float*)(ws + qkv_off);
  unsigned short* Qb = (unsigned short*)(ws + alloc((size_t)HEADS * SEQ * HD * 2));
  unsigned short* Kb = (unsigned short*)(ws + alloc((size_t)HEADS * SEQ * HD * 2));
  unsigned short* Vb = (unsigned short*)(ws + alloc((size_t)HEADS * SEQ * HD * 2));
  float* hiddenf = (float*)(ws + alloc((size_t)SIMG * DIM * 4));
  float* encf    = (float*)(ws + alloc((size_t)STXT * DIM * 4));
  // Reuse: attn output lives in the (dead) A_img+A_txt region (exactly SEQ*DIM bf16)
  unsigned short* attnb = A_img;
  // Reuse: pass-2 activations + MLP hidden live in the (dead) qkv region
  unsigned short* A2_img = (unsigned short*)(ws + qkv_off);
  unsigned short* A2_txt = A2_img + (size_t)SIMG * DIM;
  unsigned short* H_img  = A2_txt + (size_t)STXT * DIM;
  unsigned short* H_txt  = H_img + (size_t)SIMG * 4 * DIM;

  const int FF = 4 * DIM;

  // 1) silu(temb), modulation vectors (reordered into 6 blocks of DIM)
  silu_kernel<<<(DIM + 255) / 256, 256, 0, stream>>>(temb, s_temb);
  modv_kernel<<<(6 * DIM) / 256, 256, 0, stream>>>(s_temb, w_imod, b_imod, img_mod);
  modv_kernel<<<(6 * DIM) / 256, 256, 0, stream>>>(s_temb, w_tmod, b_tmod, txt_mod);

  // 2) LN + modulate (mod1: shift@0, scale@DIM) -> bf16 A
  ln_mod_kernel<<<SIMG, 256, 0, stream>>>(hs, img_mod, img_mod + DIM, A_img);
  ln_mod_kernel<<<STXT, 256, 0, stream>>>(es, txt_mod, txt_mod + DIM, A_txt);

  // 3) QKV GEMMs (txt rows 0..511, img rows 512..1535)
  gemm_bf16_kernel<<<dim3(3 * DIM / BN, STXT / BM), 256, 0, stream>>>(
      A_txt, w_aqkv, b_aqkv, STXT, 3 * DIM, DIM, 0, qkv, nullptr, nullptr, nullptr);
  gemm_bf16_kernel<<<dim3(3 * DIM / BN, SIMG / BM), 256, 0, stream>>>(
      A_img, w_qkv, b_qkv, SIMG, 3 * DIM, DIM, 0,
      qkv + (size_t)STXT * 3 * DIM, nullptr, nullptr, nullptr);

  // 4) RMSNorm + RoPE + scale -> head-major bf16 Q/K/V
  rms_rope_kernel<<<SEQ * HEADS, 64, 0, stream>>>(
      qkv, nqw, nkw, naqw, nakw, rcos, rsin, Qb, Kb, Vb);

  // 5) Flash attention -> attn bf16 [SEQ, DIM]
  attn_kernel<<<dim3(SEQ / 64, HEADS), 128, 0, stream>>>(Qb, Kb, Vb, attnb);

  // 6) out projections with gated residual (gate1 @ mod offset 2*DIM)
  gemm_bf16_kernel<<<dim3(DIM / BN, SIMG / BM), 256, 0, stream>>>(
      attnb + (size_t)STXT * DIM, w_o, b_o, SIMG, DIM, DIM, 2,
      hiddenf, nullptr, hs, img_mod + 2 * DIM);
  gemm_bf16_kernel<<<dim3(DIM / BN, STXT / BM), 256, 0, stream>>>(
      attnb, w_ao, b_ao, STXT, DIM, DIM, 2,
      encf, nullptr, es, txt_mod + 2 * DIM);

  // 7) LN + modulate pass 2 (mod2: shift@3*DIM, scale@4*DIM)
  ln_mod_kernel<<<SIMG, 256, 0, stream>>>(hiddenf, img_mod + 3 * DIM, img_mod + 4 * DIM, A2_img);
  ln_mod_kernel<<<STXT, 256, 0, stream>>>(encf, txt_mod + 3 * DIM, txt_mod + 4 * DIM, A2_txt);

  // 8) MLP1 with tanh-GELU -> bf16
  gemm_bf16_kernel<<<dim3(FF / BN, SIMG / BM), 256, 0, stream>>>(
      A2_img, w_im1, b_im1, SIMG, FF, DIM, 1, nullptr, H_img, nullptr, nullptr);
  gemm_bf16_kernel<<<dim3(FF / BN, STXT / BM), 256, 0, stream>>>(
      A2_txt, w_tm1, b_tm1, STXT, FF, DIM, 1, nullptr, H_txt, nullptr, nullptr);

  // 9) MLP2 with gated residual (gate2 @ mod offset 5*DIM) -> d_out
  gemm_bf16_kernel<<<dim3(DIM / BN, STXT / BM), 256, 0, stream>>>(
      H_txt, w_tm2, b_tm2, STXT, DIM, FF, 2,
      out, nullptr, encf, txt_mod + 5 * DIM);
  gemm_bf16_kernel<<<dim3(DIM / BN, SIMG / BM), 256, 0, stream>>>(
      H_img, w_im2, b_im2, SIMG, DIM, FF, 2,
      out + (size_t)STXT * DIM, nullptr, hiddenf, img_mod + 5 * DIM);
}
